// MultiHeadAttn1D_13554916786333
// MI455X (gfx1250) — compile-verified, matrix operand order checked by reading, not run on hardware
//
#include <hip/hip_runtime.h>
#include <hip/hip_bf16.h>
#include <stdint.h>

// ---------------------------------------------------------------------------
// Multi-head attention (N=4, H=8, T=2048, Cin=256, Ckq=Cout=512, dk=dv=64)
// bf16 WMMA flash-attention pipeline for gfx1250 with async-to-LDS staging.
// ---------------------------------------------------------------------------

typedef __attribute__((ext_vector_type(16))) __bf16 v16bf;
typedef __attribute__((ext_vector_type(8)))  __bf16 v8bf;
typedef __attribute__((ext_vector_type(8)))  float  v8f;
typedef __attribute__((ext_vector_type(4)))  int    v4i;

#define N_B   4
#define C_IN  256
#define T_LEN 2048
#define C_KQ  512
#define HEADS 8
#define DK    64
#define DV    64

// Padded LDS row strides (bf16 elements) to spread b128 reads across banks.
#define SKP 72            // K tile: 32 rows x 64 d  (stride 72, 144B = 9*16)
#define SVP 40            // V tile: 64 rows x 32 tk (stride 40,  80B = 5*16)

#define USE_ASYNC_LDS (__has_builtin(__builtin_amdgcn_global_load_async_to_lds_b128))

#if USE_ASYNC_LDS
typedef __attribute__((address_space(1))) v4i* as1_v4i_p;   // global
typedef __attribute__((address_space(3))) v4i* as3_v4i_p;   // LDS
#endif

// ---------------------------------------------------------------------------
// Fragment loader matching the CDNA5 16-bit A/B VGPR layout (ISA 7.12.2):
//   lanes 0-15 : row/col = lane,     K = 0..7  and 16..23
//   lanes 16-31: row/col = lane-16,  K = 8..15 and 24..31
// -> two contiguous 16-byte loads per lane (global_load_b128 or ds_load_b128).
// ---------------------------------------------------------------------------
__device__ __forceinline__ v16bf load_frag_bf16(const __bf16* p, int stride) {
    const int lane = threadIdx.x & 31;
    const __bf16* q = p + (size_t)(lane & 15) * stride + ((lane >> 4) << 3);
    v16bf r;
    *(v8bf*)&r       = *(const v8bf*)(q);
    *((v8bf*)&r + 1) = *(const v8bf*)(q + 16);
    return r;
}

__device__ __forceinline__ v8f wmma_bf16(v16bf a, v16bf b, v8f c) {
    return __builtin_amdgcn_wmma_f32_16x16x32_bf16(
        /*neg_a=*/false, a, /*neg_b=*/false, b,
        /*c_mod=*/(short)0, c, /*reuse_a=*/false, /*reuse_b=*/false);
}

// 16-byte global -> LDS copy: async DMA (ASYNCcnt) when available, else
// explicit load + ds_store (synchronized by the workgroup barrier).
__device__ __forceinline__ void copy_b128_to_lds(const __bf16* g, __bf16* l) {
#if USE_ASYNC_LDS
    __builtin_amdgcn_global_load_async_to_lds_b128(
        (as1_v4i_p)(g), (as3_v4i_p)(l), /*offset=*/0, /*cpol=*/0);
#else
    *(v8bf*)l = *(const v8bf*)g;
#endif
}

__device__ __forceinline__ void stage_fence() {
#if USE_ASYNC_LDS
#if __has_builtin(__builtin_amdgcn_s_wait_asynccnt)
    __builtin_amdgcn_s_wait_asynccnt(0);
#else
    asm volatile("s_wait_asynccnt 0x0" ::: "memory");
#endif
#endif
    __syncthreads();
}

// ---------------------------------------------------------------------------
// Kernel 1a: x (N,Cin,T) f32  ->  xT (N,T,Cin) bf16
// ---------------------------------------------------------------------------
__global__ __launch_bounds__(256) void k_cvt_x(const float* __restrict__ x,
                                               __bf16* __restrict__ xt) {
    int idx = blockIdx.x * 256 + threadIdx.x;            // N*Cin*T threads
    int t = idx % T_LEN;
    int c = (idx / T_LEN) % C_IN;
    int n = idx / (T_LEN * C_IN);
    xt[((size_t)n * T_LEN + t) * C_IN + c] = (__bf16)x[idx];
}

// ---------------------------------------------------------------------------
// Kernel 1b: weights f32 -> bf16 (softmax scale 1/sqrt(dk) folded into Wq)
// ---------------------------------------------------------------------------
__global__ __launch_bounds__(256) void k_cvt_w(const float* __restrict__ Wq,
                                               const float* __restrict__ Wk,
                                               const float* __restrict__ Wv,
                                               __bf16* __restrict__ wq,
                                               __bf16* __restrict__ wk,
                                               __bf16* __restrict__ wv) {
    const int WN = C_KQ * C_IN;
    int idx = blockIdx.x * 256 + threadIdx.x;            // 3*WN threads
    if (idx < WN) {
        wq[idx] = (__bf16)(Wq[idx] * 0.125f);            // 1/sqrt(64)
    } else if (idx < 2 * WN) {
        wk[idx - WN] = (__bf16)Wk[idx - WN];
    } else {
        wv[idx - 2 * WN] = (__bf16)Wv[idx - 2 * WN];
    }
}

// ---------------------------------------------------------------------------
// Kernel 2: projections.  One wave = one 16x16 tile of W(512x256) @ x(256xT).
//   mat 0 -> Qt (nh,T,dk) transposed   (scale pre-folded)
//   mat 1 -> Kt (nh,T,dk) transposed
//   mat 2 -> V  (nh,dv,T) natural
// ---------------------------------------------------------------------------
__global__ __launch_bounds__(256) void k_proj(const __bf16* __restrict__ xt,
                                              const __bf16* __restrict__ wq,
                                              const __bf16* __restrict__ wk,
                                              const __bf16* __restrict__ wv,
                                              __bf16* __restrict__ Qt,
                                              __bf16* __restrict__ Kt,
                                              __bf16* __restrict__ Vv) {
    const int wave = threadIdx.x >> 5;
    const int lane = threadIdx.x & 31;
    int tile = blockIdx.x * 8 + wave;     // N*3*(512/16)*(T/16) tiles total
    const int ncol = tile & 127;  tile >>= 7;   // T/16  = 128
    const int mrow = tile & 31;   tile >>= 5;   // 512/16 = 32
    const int mat  = tile % 3;
    const int n    = tile / 3;

    const __bf16* W  = (mat == 0) ? wq : (mat == 1) ? wk : wv;
    const __bf16* wb = W + (size_t)mrow * 16 * C_IN;
    const __bf16* xb = xt + ((size_t)n * T_LEN + (size_t)ncol * 16) * C_IN;

    v8f acc = {};
#pragma unroll
    for (int k0 = 0; k0 < C_IN; k0 += 32) {
        v16bf a = load_frag_bf16(wb + k0, C_IN);   // 16 rows of W
        v16bf b = load_frag_bf16(xb + k0, C_IN);   // 16 timesteps of xT
        acc = wmma_bf16(a, b, acc);
    }

    // C/D layout: lane<16 -> col=lane, rows r=0..7 ; lane>=16 -> rows 8..15
    const int moff = (lane >> 4) << 3;             // 0 or 8
    const int g0   = mrow * 16 + moff;             // first of 8 output rows
    const int t    = ncol * 16 + (lane & 15);
    const int head = g0 >> 6;
    const int d0   = g0 & 63;
    const int nh   = n * HEADS + head;

    if (mat < 2) {
        __bf16* dst = (mat == 0) ? Qt : Kt;        // (nh, T, dk) -> one b128
        v8bf pk;
#pragma unroll
        for (int r = 0; r < 8; ++r) pk[r] = (__bf16)acc[r];
        *(v8bf*)(dst + ((size_t)nh * T_LEN + t) * DK + d0) = pk;
    } else {                                       // V natural (nh, dv, T)
        size_t base = ((size_t)nh * DV + d0) * T_LEN + t;
#pragma unroll
        for (int r = 0; r < 8; ++r) Vv[base + (size_t)r * T_LEN] = (__bf16)acc[r];
    }
}

// ---------------------------------------------------------------------------
// Kernel 3: flash attention, block-cooperative.
// One block = one (n,h) x 128 query columns (8 waves x 16 each).
// Per iteration over 32 keys:
//   - async-stage next K (32x64) and V (64x32) tiles into LDS (double buffer)
//   - 4 WMMA for S, online softmax (one shfl_xor(16) per reduction),
//     lane-local repack of P into a bf16 B-fragment, 4 WMMA for O.
// Global traffic: 8 KB/iter/block instead of 64 KB (waves share tiles).
// ---------------------------------------------------------------------------
__global__ __launch_bounds__(256) void k_attn(const __bf16* __restrict__ Qt,
                                              const __bf16* __restrict__ Kt,
                                              const __bf16* __restrict__ Vv,
                                              float* __restrict__ out) {
    __shared__ __bf16 sK[2][32 * SKP];
    __shared__ __bf16 sV[2][64 * SVP];

    const int tid  = threadIdx.x;
    const int wave = tid >> 5;
    const int lane = tid & 31;
    const int nh   = blockIdx.y;                          // 0..31
    const int tq   = blockIdx.x * 128 + wave * 16;        // query tile base

    const __bf16* qp = Qt + ((size_t)nh * T_LEN + tq) * DK;
    const __bf16* kp = Kt + (size_t)nh * T_LEN * DK;
    const __bf16* vp = Vv + (size_t)nh * DV * T_LEN;

    // Staging addresses for this thread (one b128 each for K and V tiles).
    const int krow = tid >> 3, kcol = (tid & 7) << 3;     // 32 rows x 64 d
    const int vrow = tid >> 2, vcol = (tid & 3) << 3;     // 64 rows x 32 tk

    // Stage tile 0.
    copy_b128_to_lds(kp + (size_t)krow * DK + kcol, &sK[0][krow * SKP + kcol]);
    copy_b128_to_lds(vp + (size_t)vrow * T_LEN + vcol, &sV[0][vrow * SVP + vcol]);

    // Q B-fragments, resident for the whole key loop (scale pre-folded).
    const v16bf bq0 = load_frag_bf16(qp + 0,  DK);        // d = 0..31
    const v16bf bq1 = load_frag_bf16(qp + 32, DK);        // d = 32..63

    v8f o0 = {}, o1 = {}, o2 = {}, o3 = {};
    float m_i = -3.0e38f;
    float l_i = 0.0f;

    stage_fence();                                        // tile 0 visible

    for (int tk = 0; tk < T_LEN; tk += 32) {
        const int buf = (tk >> 5) & 1;

        // Kick off the next tile's DMA before computing on this one.
        if (tk + 32 < T_LEN) {
            const int nb = buf ^ 1;
            copy_b128_to_lds(kp + (size_t)(tk + 32) * DK + krow * DK + kcol,
                             &sK[nb][krow * SKP + kcol]);
            copy_b128_to_lds(vp + (size_t)vrow * T_LEN + (tk + 32) + vcol,
                             &sV[nb][vrow * SVP + vcol]);
        }

        const __bf16* kb = sK[buf];
        const __bf16* vb = sV[buf];

        // K A-fragments from LDS (rows tk..+15 and +16..+31, both d-halves)
        v16bf ak00 = load_frag_bf16(kb, SKP);
        v16bf ak01 = load_frag_bf16(kb + 32, SKP);
        v16bf ak10 = load_frag_bf16(kb + 16 * SKP, SKP);
        v16bf ak11 = load_frag_bf16(kb + 16 * SKP + 32, SKP);

        v8f s0 = {};
        s0 = wmma_bf16(ak00, bq0, s0);
        s0 = wmma_bf16(ak01, bq1, s0);
        v8f s1 = {};
        s1 = wmma_bf16(ak10, bq0, s1);
        s1 = wmma_bf16(ak11, bq1, s1);

        // ---- online softmax over the key axis (M axis of the fragments) ---
        float mloc = m_i;
#pragma unroll
        for (int r = 0; r < 8; ++r)
            mloc = fmaxf(mloc, fmaxf(s0[r], s1[r]));
        // lanes L and L+16 hold the two row-halves of the same tq column
        const float mnew  = fmaxf(mloc, __shfl_xor(mloc, 16));
        const float alpha = __expf(m_i - mnew);

        float p0[8], p1[8], ps = 0.0f;
#pragma unroll
        for (int r = 0; r < 8; ++r) {
            p0[r] = __expf(s0[r] - mnew);
            p1[r] = __expf(s1[r] - mnew);
            ps += p0[r] + p1[r];
        }
        ps += __shfl_xor(ps, 16);
        l_i = l_i * alpha + ps;
        m_i = mnew;

#pragma unroll
        for (int r = 0; r < 8; ++r) {
            o0[r] *= alpha; o1[r] *= alpha; o2[r] *= alpha; o3[r] *= alpha;
        }

        // P as a bf16 B-fragment: purely lane-local repack of s0/s1.
        v16bf bp;
#pragma unroll
        for (int r = 0; r < 8; ++r) {
            bp[r]     = (__bf16)p0[r];   // K = tk .. tk+15 half
            bp[8 + r] = (__bf16)p1[r];   // K = tk+16 .. tk+31 half
        }

        // V A-fragments from LDS: four 16-row dv tiles, K = 32 keys.
        v16bf av0 = load_frag_bf16(vb + 0 * 16 * SVP, SVP);
        v16bf av1 = load_frag_bf16(vb + 1 * 16 * SVP, SVP);
        v16bf av2 = load_frag_bf16(vb + 2 * 16 * SVP, SVP);
        v16bf av3 = load_frag_bf16(vb + 3 * 16 * SVP, SVP);

        o0 = wmma_bf16(av0, bp, o0);
        o1 = wmma_bf16(av1, bp, o1);
        o2 = wmma_bf16(av2, bp, o2);
        o3 = wmma_bf16(av3, bp, o3);

        if (tk + 32 < T_LEN)
            stage_fence();               // next tile ready, this one reusable
    }

    const float inv = 1.0f / l_i;
    const int n  = nh >> 3;
    const int h  = nh & 7;
    const int t  = tq + (lane & 15);
    const int d0 = (lane >> 4) << 3;

#pragma unroll
    for (int dvt = 0; dvt < 4; ++dvt) {
        const v8f& o = (dvt == 0) ? o0 : (dvt == 1) ? o1 : (dvt == 2) ? o2 : o3;
        size_t base = ((size_t)n * C_KQ + h * DV + dvt * 16 + d0) * T_LEN + t;
#pragma unroll
        for (int r = 0; r < 8; ++r)
            out[base + (size_t)r * T_LEN] = o[r] * inv;
    }
}

// ---------------------------------------------------------------------------
// Host launcher. Workspace layout (bf16 elements):
//   xT: N*T*Cin | wq,wk,wv: 3*512*256 | Qt,Kt: N*H*T*64 | V: N*H*64*T
// total ~30 MB (28.75 MiB).
// ---------------------------------------------------------------------------
extern "C" void kernel_launch(void* const* d_in, const int* in_sizes, int n_in,
                              void* d_out, int out_size, void* d_ws, size_t ws_size,
                              hipStream_t stream) {
    (void)in_sizes; (void)n_in; (void)out_size; (void)ws_size;

    const float* x  = (const float*)d_in[0];
    const float* Wq = (const float*)d_in[1];
    const float* Wk = (const float*)d_in[2];
    const float* Wv = (const float*)d_in[3];
    float* out = (float*)d_out;

    __bf16* ws = (__bf16*)d_ws;
    const size_t nXT = (size_t)N_B * T_LEN * C_IN;        // 2 Mi elems
    const size_t nW  = (size_t)C_KQ * C_IN;               // 128 Ki elems
    const size_t nQ  = (size_t)N_B * HEADS * T_LEN * DK;  // 4 Mi elems

    __bf16* xt = ws;
    __bf16* wq = xt + nXT;
    __bf16* wk = wq + nW;
    __bf16* wv = wk + nW;
    __bf16* Qt = wv + nW;
    __bf16* Kt = Qt + nQ;
    __bf16* Vv = Kt + nQ;

    // 1) f32 -> bf16 conversions (+ transpose of x, scale fold into Wq)
    k_cvt_x<<<(N_B * C_IN * T_LEN) / 256, 256, 0, stream>>>(x, xt);
    k_cvt_w<<<(3 * C_KQ * C_IN) / 256, 256, 0, stream>>>(Wq, Wk, Wv, wq, wk, wv);

    // 2) Q/K/V projections: N*3*(512/16)*(T/16) = 49152 wave-tiles, 8/block
    k_proj<<<49152 / 8, 256, 0, stream>>>(xt, wq, wk, wv, Qt, Kt, Vv);

    // 3) flash attention: 32 (n,h) pairs x (T/128 = 16) blocks each
    dim3 agrid(T_LEN / 128, N_B * HEADS);                 // (16, 32)
    k_attn<<<agrid, 256, 0, stream>>>(Qt, Kt, Vv, out);
}
